// Token2Word_86431921865311
// MI455X (gfx1250) — matrix operand order repaired and checked by reading, not compiled, not run on hardware
//
#include <hip/hip_runtime.h>
#include <hip/hip_bf16.h>

// Token->word mean pooling as a batched WMMA GEMM:
//   E[b] = P[b] (W x S, 0/1 band rows, generated in-register) * Hs[b] (S x H)
// f32 accumulate via v_wmma_f32_16x16x32_f16; per-row scale by valid/len in f32.
//
// Round-6 changes vs round 5:
//  - epilogue uses v_rcp_f32 (1-ulp) instead of exact-div fixup chains
//  - A-fragment bit expansion via pre-spread masks: 3 VALU per packed pair

typedef __attribute__((ext_vector_type(16))) _Float16 v16h;
typedef __attribute__((ext_vector_type(8)))  float    v8f;

namespace {
constexpr int Bn = 16;
constexpr int Sn = 4096;
constexpr int Hn = 1024;
constexpr int Wn = 1024;

constexpr int BM = 128;        // words per workgroup tile
constexpr int BN = 128;        // hidden cols per workgroup tile
constexpr int BK = 32;         // K step (matches 16x16x32 f16 WMMA)
constexpr int LDS_ROW_B = 80;  // bytes per n-row: 32 f16 (64B) + 16B pad (bank-spread)
constexpr int LDS_BUF_B = BN * LDS_ROW_B;   // 10240 B per buffer

__device__ __forceinline__ unsigned ones_from(int bit) {
    // 32-bit value with bits >= clamp(bit,0,32) set
    int c = bit < 0 ? 0 : (bit > 32 ? 32 : bit);
    return (unsigned)(~0ull << (unsigned)c);
}
}

__global__ __launch_bounds__(256)
__attribute__((amdgpu_waves_per_eu(2, 4)))
void t2w_wmma_kernel(
    const float* __restrict__ hs,     // (B, S, H) f32
    const int*   __restrict__ widx,   // (B, W, 2) i32
    const int*   __restrict__ mwl,    // (B,) i32
    float*       __restrict__ out)    // (B, W, H) f32
{
    __shared__ __align__(16) unsigned char ldsB[2 * LDS_BUF_B];  // 20 KB

    const int b    = blockIdx.z;
    const int w0   = blockIdx.y * BM;
    const int h0   = blockIdx.x * BN;
    const int tid  = threadIdx.x;
    const int lane = tid & 31;
    const int wv   = tid >> 5;        // wave 0..7
    const int half = lane >> 4;       // 0: lanes 0-15, 1: lanes 16-31
    const int l15  = lane & 15;

    // wave -> 32x64 output strip: rows [w0+32*(wv&3), +32), cols [h0+64*(wv>>2), +64)
    const int mrow = (wv & 3) * 32;   // word-row offset of this wave's strip
    const int ncol = (wv >> 2) * 64;  // hidden-col offset of this wave's strip

    // --- per-lane A-row spans for the 2 M-subtiles (A row m = lane&15) ---
    int sA[2], eA1[2];
    #pragma unroll
    for (int mh = 0; mh < 2; ++mh) {
        const int aw = w0 + mrow + mh * 16 + l15;
        sA[mh]  = widx[((size_t)b * Wn + aw) * 2 + 0];
        eA1[mh] = widx[((size_t)b * Wn + aw) * 2 + 1] + 1;
    }

    const float* hbase = hs + (size_t)b * Sn * Hn;

    // --- loader mapping: thread t loads a 16-deep K column at fixed h ---
    const int ln = tid & 127;             // n within tile: 0..127
    const int lk = (tid >> 7) * 16;       // 0 or 16 (K sub-block)
    const size_t wrOff = (size_t)ln * LDS_ROW_B + (size_t)lk * 2;  // 16B aligned
    const float* gcol = hbase + (size_t)lk * Hn + (h0 + ln);

    v8f acc[2][4] = {};   // [mh][nt] 16x16 f32 accumulators (64 VGPRs)

    // stage tile 0 into registers (128B-coalesced per wave per load)
    float f[16];
    #pragma unroll
    for (int i = 0; i < 16; ++i) f[i] = gcol[(size_t)i * Hn];

    int p = 0;
    for (int kk = 0; kk < Sn; kk += BK) {
        // ---- convert staged f32 -> f16, store to LDS buffer p ([n][k]) ----
        union { _Float16 h[16]; uint4 q[2]; } pk;
        #pragma unroll
        for (int i = 0; i < 16; ++i) pk.h[i] = (_Float16)f[i];
        unsigned char* wr = ldsB + p * LDS_BUF_B + wrOff;
        *(uint4*)(wr)      = pk.q[0];
        *(uint4*)(wr + 16) = pk.q[1];

        __syncthreads();   // single barrier per iteration (double-buffered LDS)

        // ---- stage NEXT tile's f32 from global (latency crosses the WMMAs) ----
        if (kk + BK < Sn) {
            const float* g = gcol + (size_t)(kk + BK) * Hn;
            #pragma unroll
            for (int i = 0; i < 16; ++i) f[i] = g[(size_t)i * Hn];
            if (kk + 2 * BK < Sn)
                __builtin_prefetch(gcol + (size_t)(kk + 2 * BK) * Hn, 0, 3);
        }

        // ---- preload the 4 B fragments (one ds clause) ----
        const unsigned char* rdBase =
            ldsB + p * LDS_BUF_B + (size_t)(ncol + l15) * LDS_ROW_B + (half << 4);
        union BF { uint4 q[2]; v16h h; } Bf[4];
        #pragma unroll
        for (int nt = 0; nt < 4; ++nt) {
            const unsigned char* bp = rdBase + nt * 16 * LDS_ROW_B;
            Bf[nt].q[0] = *(const uint4*)(bp);        // ds_load_b128
            Bf[nt].q[1] = *(const uint4*)(bp + 32);   // ds_load_b128
        }

        // ---- generate 2 A fragments via window mask + spread-bit expansion ----
        // bit i of mask <=> token kk+i inside [s, e].
        // A VGPR v holds K pair at offset half*8 + (v&3)*2 + (v>>2)*16.
        union AF { unsigned int u[8]; v16h h; } A[2];
        #pragma unroll
        for (int mh = 0; mh < 2; ++mh) {
            const unsigned mask =
                ones_from(sA[mh] - kk) ^ ones_from(eA1[mh] - kk);
            const unsigned msh = mask >> (half << 3);  // lane-half aligned
            const unsigned m16[2] = { msh & 0xFFFFu, msh >> 16 };
            // sp bit c -> token bit c ; sp bit c+16 -> token bit c+1
            const unsigned sp[2] = { m16[0] | (m16[0] << 15),
                                     m16[1] | (m16[1] << 15) };
            #pragma unroll
            for (int v = 0; v < 8; ++v) {
                const unsigned tt2 = (sp[v >> 2] >> ((v & 3) << 1)) & 0x10001u;
                A[mh].u[v] = tt2 * 0x3C00u;   // packed f16 {0.0, 1.0}
            }
        }

        // ---- 8 WMMAs; each B fragment feeds 2 WMMAs ----
        #pragma unroll
        for (int nt = 0; nt < 4; ++nt) {
            #pragma unroll
            for (int mh = 0; mh < 2; ++mh) {
                acc[mh][nt] = __builtin_amdgcn_wmma_f32_16x16x32_f16(
                    false, A[mh].h, false, Bf[nt].h, (short)0, acc[mh][nt],
                    false, false);
            }
        }

        p ^= 1;
    }

    // ---- epilogue: scale rows by valid/len in f32, store ----
    // C layout: VGPR j, lane -> row (half*8 + j), col l15
    const int maxw = mwl[b];
    float scale[2][8];
    #pragma unroll
    for (int mh = 0; mh < 2; ++mh) {
        #pragma unroll
        for (int j = 0; j < 8; ++j) {
            const int w = w0 + mrow + mh * 16 + (half << 3) + j;
            const int ss = widx[((size_t)b * Wn + w) * 2 + 0];
            const int ee = widx[((size_t)b * Wn + w) * 2 + 1];
            const float r = __builtin_amdgcn_rcpf((float)(ee - ss + 1));
            scale[mh][j] = (w < maxw) ? r : 0.0f;
        }
    }

    #pragma unroll
    for (int mh = 0; mh < 2; ++mh) {
        #pragma unroll
        for (int nt = 0; nt < 4; ++nt) {
            #pragma unroll
            for (int j = 0; j < 8; ++j) {
                const int w = w0 + mrow + mh * 16 + (half << 3) + j;
                out[((size_t)b * Wn + w) * Hn + h0 + ncol + nt * 16 + l15] =
                    acc[mh][nt][j] * scale[mh][j];
            }
        }
    }
}

__global__ __launch_bounds__(256) void t2w_mask_kernel(
    const int* __restrict__ mwl, long long* __restrict__ masks)
{
    const int i = blockIdx.x * 256 + threadIdx.x;   // over B*W
    if (i < Bn * Wn) {
        const int b = i >> 10;        // W = 1024
        const int w = i & (Wn - 1);
        masks[i] = (w < mwl[b]) ? 1LL : 0LL;
    }
}

extern "C" void kernel_launch(void* const* d_in, const int* in_sizes, int n_in,
                              void* d_out, int out_size, void* d_ws, size_t ws_size,
                              hipStream_t stream) {
    (void)in_sizes; (void)n_in; (void)out_size; (void)d_ws; (void)ws_size;
    const float* hs  = (const float*)d_in[0];
    const int* widx  = (const int*)d_in[1];
    const int* mwl   = (const int*)d_in[2];
    float* out       = (float*)d_out;
    long long* masks = (long long*)(out + (size_t)Bn * Wn * Hn);

    dim3 grid(Hn / BN, Wn / BM, Bn);   // (8, 8, 16)
    t2w_wmma_kernel<<<grid, 256, 0, stream>>>(hs, widx, mwl, out);

    t2w_mask_kernel<<<(Bn * Wn + 255) / 256, 256, 0, stream>>>(mwl, masks);
}
